// FeedForwardBlock_64372969832565
// MI455X (gfx1250) — 16-bit and 8-bit matrix operand order repaired and checked by reading, compiled, not run on hardware
//
#include <hip/hip_runtime.h>

#define B_   4
#define S_   4096
#define D_   1024
#define FF_  4096
#define C_   256
#define M_   (B_ * S_)
#define EPS_ 1e-6f

typedef __attribute__((ext_vector_type(16))) __bf16 v16bf;
typedef __attribute__((ext_vector_type(8)))  __bf16 v8bf;
typedef __attribute__((ext_vector_type(4)))  __bf16 v4bf;
typedef __attribute__((ext_vector_type(8)))  float  v8f;
typedef __attribute__((ext_vector_type(8)))  int    v8i;
typedef __attribute__((ext_vector_type(4)))  int    v4i;
typedef __attribute__((ext_vector_type(2)))  int    v2i;

#if __has_builtin(__builtin_amdgcn_global_load_async_to_lds_b128) && \
    __has_builtin(__builtin_amdgcn_s_wait_asynccnt)
#define ASYNC_LDS 1
typedef __attribute__((address_space(1))) v4i g_v4i;
typedef __attribute__((address_space(3))) v4i l_v4i;
#else
#define ASYNC_LDS 0
#endif

__device__ __forceinline__ void async_copy16(void* lds_dst, const void* gsrc) {
#if ASYNC_LDS
  __builtin_amdgcn_global_load_async_to_lds_b128((g_v4i*)gsrc, (l_v4i*)lds_dst, 0, 0);
#else
  *(v4i*)lds_dst = *(const v4i*)gsrc;
#endif
}

__device__ __forceinline__ void async_wait_all() {
#if ASYNC_LDS
  __builtin_amdgcn_s_wait_asynccnt(0);
#endif
}

__global__ __launch_bounds__(256) void k_scale(const float* __restrict__ cond,
                                               const float* __restrict__ nw,
                                               float* __restrict__ scale) {
  int idx = blockIdx.x * 256 + threadIdx.x;
  int b = idx / D_, d = idx - b * D_;
  float s = 1.0f;
  const float* cp = cond + b * C_;
  const float* wp = nw + d * C_;
#pragma unroll 4
  for (int c = 0; c < C_; ++c) s += cp[c] * wp[c];
  scale[idx] = s;
}

__global__ __launch_bounds__(256) void k_rmsnorm(const float* __restrict__ x,
                                                 const float* __restrict__ scale,
                                                 __bf16* __restrict__ h) {
  int m = blockIdx.x;
  int b = m / S_;
  const float4* xr = (const float4*)(x + (size_t)m * D_);
  float4 v = xr[threadIdx.x];
  float ss = v.x * v.x + v.y * v.y + v.z * v.z + v.w * v.w;
  __shared__ float red[256];
  red[threadIdx.x] = ss;
  __syncthreads();
  for (int s = 128; s > 0; s >>= 1) {
    if (threadIdx.x < s) red[threadIdx.x] += red[threadIdx.x + s];
    __syncthreads();
  }
  float inv = rsqrtf(red[0] * (1.0f / D_) + EPS_);
  const float4* sp = (const float4*)(scale + b * D_);
  float4 s4 = sp[threadIdx.x];
  v4bf o;
  o[0] = (__bf16)(v.x * s4.x * inv);
  o[1] = (__bf16)(v.y * s4.y * inv);
  o[2] = (__bf16)(v.z * s4.z * inv);
  o[3] = (__bf16)(v.w * s4.w * inv);
  ((v4bf*)(h + (size_t)m * D_))[threadIdx.x] = o;
}

__global__ __launch_bounds__(256) void k_cast_upw(const float* __restrict__ w,
                                                  __bf16* __restrict__ wb) {
  int i = blockIdx.x * 256 + threadIdx.x;
  float4 v = ((const float4*)w)[i];
  v4bf o = { (__bf16)v.x, (__bf16)v.y, (__bf16)v.z, (__bf16)v.w };
  ((v4bf*)wb)[i] = o;
}

__global__ __launch_bounds__(256) void k_wabs(const float* __restrict__ w,
                                              float* __restrict__ acc) {
  int i = blockIdx.x * 256 + threadIdx.x;
  float4 v = ((const float4*)w)[i];
  float s = fabsf(v.x) + fabsf(v.y) + fabsf(v.z) + fabsf(v.w);
  __shared__ float red[256];
  red[threadIdx.x] = s;
  __syncthreads();
  for (int t = 128; t > 0; t >>= 1) {
    if (threadIdx.x < t) red[threadIdx.x] += red[threadIdx.x + t];
    __syncthreads();
  }
  if (threadIdx.x == 0) atomicAdd(acc, red[0]);
}

__global__ __launch_bounds__(256) void k_wquant(const float* __restrict__ w,
                                                const float* __restrict__ acc,
                                                signed char* __restrict__ wq) {
  float mean = acc[0] * (1.0f / ((float)D_ * (float)FF_));
  float sc = 1.0f / fmaxf(mean, 1e-5f);
  int i = blockIdx.x * 256 + threadIdx.x;
  float4 v = ((const float4*)w)[i];
  int q0 = (int)fminf(fmaxf(rintf(v.x * sc), -1.0f), 1.0f);
  int q1 = (int)fminf(fmaxf(rintf(v.y * sc), -1.0f), 1.0f);
  int q2 = (int)fminf(fmaxf(rintf(v.z * sc), -1.0f), 1.0f);
  int q3 = (int)fminf(fmaxf(rintf(v.w * sc), -1.0f), 1.0f);
  ((int*)wq)[i] = (q0 & 255) | ((q1 & 255) << 8) | ((q2 & 255) << 16) | ((q3 & 255) << 24);
}

__device__ __forceinline__ void stage_b1(__bf16* ldsB, const __bf16* __restrict__ w,
                                         int f0, int kk, int tid) {
  int r  = tid >> 1;
  int ko = (tid & 1) * 16;
  int wrow = (r < 64) ? (f0 + r) : (FF_ + f0 + (r - 64));
  const __bf16* src = w + (size_t)wrow * D_ + kk + ko;
  __bf16* dst = ldsB + r * 32 + ko;
  async_copy16(dst, src);
  async_copy16(dst + 8, src + 8);
}

__global__ __launch_bounds__(256) void k_gemm1_geglu(const __bf16* __restrict__ h,
                                                     const __bf16* __restrict__ w,
                                                     float* __restrict__ hg) {
  __shared__ __bf16 ldsB[2][128 * 32];
  const int lane = threadIdx.x & 31;
  const int wave = threadIdx.x >> 5;
  const int m0 = blockIdx.x * 128 + wave * 16;
  const int f0 = blockIdx.y * 64;

  v8f accv[4], accg[4];
#pragma unroll
  for (int i = 0; i < 4; ++i) {
#pragma unroll
    for (int j = 0; j < 8; ++j) { accv[i][j] = 0.0f; accg[i][j] = 0.0f; }
  }

  const int nlo = lane & 15;
  const int ahi = (lane < 16) ? 0 : 16;
  const int bhi = (lane < 16) ? 0 : 16;

  stage_b1(ldsB[0], w, f0, 0, threadIdx.x);
  async_wait_all();
  __syncthreads();

  for (int kk = 0; kk < D_; kk += 32) {
    const int cur = (kk >> 5) & 1;
    if (kk + 32 < D_)
      stage_b1(ldsB[cur ^ 1], w, f0, kk + 32, threadIdx.x);

    v16bf a;
    {
      const __bf16* base = h + (size_t)(m0 + nlo) * D_ + kk;
      v8bf a0 = *(const v8bf*)(base + ahi);
      v8bf a1 = *(const v8bf*)(base + ahi + 8);
#pragma unroll
      for (int j = 0; j < 8; ++j) { a[j] = a0[j]; a[j + 8] = a1[j]; }
    }
    if (kk + 32 < D_)
      __builtin_prefetch(h + (size_t)(m0 + nlo) * D_ + kk + 32, 0, 3);

    v16bf bv[4], bg[4];
#pragma unroll
    for (int i = 0; i < 4; ++i) {
      const v8bf* pv = (const v8bf*)(&ldsB[cur][(16 * i + nlo) * 32 + bhi]);
      const v8bf* pg = (const v8bf*)(&ldsB[cur][(64 + 16 * i + nlo) * 32 + bhi]);
      v8bf b0 = pv[0], b1 = pv[1];
      v8bf g0 = pg[0], g1 = pg[1];
#pragma unroll
      for (int j = 0; j < 8; ++j) {
        bv[i][j] = b0[j]; bv[i][j + 8] = b1[j];
        bg[i][j] = g0[j]; bg[i][j + 8] = g1[j];
      }
    }
#pragma unroll
    for (int i = 0; i < 4; ++i) {
      accv[i] = __builtin_amdgcn_wmma_f32_16x16x32_bf16(false, a, false, bv[i],
                                                        (short)0, accv[i], false, false);
      accg[i] = __builtin_amdgcn_wmma_f32_16x16x32_bf16(false, a, false, bg[i],
                                                        (short)0, accg[i], false, false);
    }
    async_wait_all();
    __syncthreads();
  }

  const int mo = (lane < 16) ? 0 : 8;
#pragma unroll
  for (int i = 0; i < 4; ++i) {
#pragma unroll
    for (int j = 0; j < 8; ++j) {
      float val = accv[i][j];
      float g   = accg[i][j];
      float ge  = 0.5f * g * (1.0f + erff(g * 0.70710678118654752f));
      int m = m0 + j + mo;
      int f = f0 + 16 * i + nlo;
      hg[(size_t)m * FF_ + f] = val * ge;
    }
  }
}

__global__ __launch_bounds__(256) void k_rowquant(const float* __restrict__ hg,
                                                  signed char* __restrict__ hq,
                                                  float* __restrict__ inv_a) {
  int m = blockIdx.x;
  const float4* row = (const float4*)(hg + (size_t)m * FF_);
  float4 v[4];
  float amax = 0.0f;
#pragma unroll
  for (int i = 0; i < 4; ++i) {
    v[i] = row[threadIdx.x + 256 * i];
    amax = fmaxf(amax, fmaxf(fmaxf(fabsf(v[i].x), fabsf(v[i].y)),
                             fmaxf(fabsf(v[i].z), fabsf(v[i].w))));
  }
  __shared__ float red[256];
  red[threadIdx.x] = amax;
  __syncthreads();
  for (int s = 128; s > 0; s >>= 1) {
    if (threadIdx.x < s) red[threadIdx.x] = fmaxf(red[threadIdx.x], red[threadIdx.x + s]);
    __syncthreads();
  }
  float amx = fmaxf(red[0], 1e-5f);
  float asc = 127.0f / amx;
  if (threadIdx.x == 0) inv_a[m] = amx * (1.0f / 127.0f);
  int* out = (int*)(hq + (size_t)m * FF_);
#pragma unroll
  for (int i = 0; i < 4; ++i) {
    int q0 = (int)fminf(fmaxf(rintf(v[i].x * asc), -128.0f), 127.0f);
    int q1 = (int)fminf(fmaxf(rintf(v[i].y * asc), -128.0f), 127.0f);
    int q2 = (int)fminf(fmaxf(rintf(v[i].z * asc), -128.0f), 127.0f);
    int q3 = (int)fminf(fmaxf(rintf(v[i].w * asc), -128.0f), 127.0f);
    out[threadIdx.x + 256 * i] =
        (q0 & 255) | ((q1 & 255) << 8) | ((q2 & 255) << 16) | ((q3 & 255) << 24);
  }
}

__device__ __forceinline__ void stage_b2(signed char* ldsW, const signed char* __restrict__ wq,
                                         int d0, int kk, int tid) {
  int r  = tid >> 2;
  int ko = (tid & 3) * 16;
  async_copy16(ldsW + r * 64 + ko, wq + (size_t)(d0 + r) * FF_ + kk + ko);
}

__global__ __launch_bounds__(256) void k_gemm2_bitlinear(const signed char* __restrict__ hq,
                                                         const signed char* __restrict__ wq,
                                                         const float* __restrict__ inv_a,
                                                         const float* __restrict__ wsum,
                                                         const float* __restrict__ x,
                                                         float* __restrict__ out) {
  __shared__ signed char ldsW[2][64 * 64];
  const int lane = threadIdx.x & 31;
  const int wave = threadIdx.x >> 5;
  const int m0 = blockIdx.x * 128 + wave * 16;
  const int d0 = blockIdx.y * 64;

  v8i acc[4];
#pragma unroll
  for (int i = 0; i < 4; ++i)
#pragma unroll
    for (int j = 0; j < 8; ++j) acc[i][j] = 0;

  const int nlo = lane & 15;
  const int ahi = (lane < 16) ? 0 : 16;
  const int bhi = (lane < 16) ? 0 : 16;
  const float wdeq = fmaxf(wsum[0] * (1.0f / ((float)D_ * (float)FF_)), 1e-5f);

  stage_b2(ldsW[0], wq, d0, 0, threadIdx.x);
  async_wait_all();
  __syncthreads();

  for (int kk = 0; kk < FF_; kk += 64) {
    const int cur = (kk >> 6) & 1;
    if (kk + 64 < FF_)
      stage_b2(ldsW[cur ^ 1], wq, d0, kk + 64, threadIdx.x);

    v8i a;
    {
      const signed char* p = hq + (size_t)(m0 + nlo) * FF_ + kk + ahi;
      v4i q0 = *(const v4i*)(p);
      v4i q1 = *(const v4i*)(p + 32);
      a[0] = q0[0]; a[1] = q0[1]; a[2] = q0[2]; a[3] = q0[3];
      a[4] = q1[0]; a[5] = q1[1]; a[6] = q1[2]; a[7] = q1[3];
    }
    if (kk + 64 < FF_)
      __builtin_prefetch(hq + (size_t)(m0 + nlo) * FF_ + kk + 64, 0, 3);

    v8i b[4];
#pragma unroll
    for (int i = 0; i < 4; ++i) {
      const signed char* p = &ldsW[cur][(16 * i + nlo) * 64 + bhi];
      v4i b0 = *(const v4i*)(p);
      v4i b1 = *(const v4i*)(p + 32);
      b[i][0] = b0[0]; b[i][1] = b0[1]; b[i][2] = b0[2]; b[i][3] = b0[3];
      b[i][4] = b1[0]; b[i][5] = b1[1]; b[i][6] = b1[2]; b[i][7] = b1[3];
    }
#pragma unroll
    for (int i = 0; i < 4; ++i) {
      acc[i] = __builtin_amdgcn_wmma_i32_16x16x64_iu8(true, a, true, b[i], acc[i],
                                                      false, false);
    }
    async_wait_all();
    __syncthreads();
  }

  const int mo = (lane < 16) ? 0 : 8;
#pragma unroll
  for (int i = 0; i < 4; ++i) {
#pragma unroll
    for (int j = 0; j < 8; ++j) {
      int m = m0 + j + mo;
      int d = d0 + 16 * i + nlo;
      float y = (float)acc[i][j] * (inv_a[m] * wdeq) + x[(size_t)m * D_ + d];
      out[(size_t)m * D_ + d] = y;
    }
  }
}

extern "C" void kernel_launch(void* const* d_in, const int* in_sizes, int n_in,
                              void* d_out, int out_size, void* d_ws, size_t ws_size,
                              hipStream_t stream) {
  const float* x      = (const float*)d_in[0];
  const float* cond   = (const float*)d_in[1];
  const float* norm_w = (const float*)d_in[2];
  const float* up_w   = (const float*)d_in[3];
  const float* down_w = (const float*)d_in[4];
  float* out = (float*)d_out;

  char* ws = (char*)d_ws;
  size_t o_scale = 0;
  size_t o_wsum  = o_scale + (size_t)B_ * D_ * 4;
  size_t o_inva  = o_wsum + 256;
  size_t o_h     = o_inva + (size_t)M_ * 4;
  size_t o_upw   = o_h + (size_t)M_ * D_ * 2;
  size_t o_wq    = o_upw + (size_t)2 * FF_ * D_ * 2;
  size_t o_hq    = o_wq + (size_t)D_ * FF_;
  size_t o_hg    = o_hq + (size_t)M_ * FF_;

  float*       scale  = (float*)(ws + o_scale);
  float*       wsum   = (float*)(ws + o_wsum);
  float*       inv_a  = (float*)(ws + o_inva);
  __bf16*      hbf    = (__bf16*)(ws + o_h);
  __bf16*      upw_bf = (__bf16*)(ws + o_upw);
  signed char* wq     = (signed char*)(ws + o_wq);
  signed char* hq     = (signed char*)(ws + o_hq);
  float*       hg     = (float*)(ws + o_hg);

  (void)hipMemsetAsync(wsum, 0, sizeof(float), stream);

  k_scale<<<(B_ * D_) / 256, 256, 0, stream>>>(cond, norm_w, scale);
  k_rmsnorm<<<M_, 256, 0, stream>>>(x, scale, hbf);
  k_cast_upw<<<(2 * FF_ * D_) / (4 * 256), 256, 0, stream>>>(up_w, upw_bf);
  k_wabs<<<(D_ * FF_) / (4 * 256), 256, 0, stream>>>(down_w, wsum);
  k_wquant<<<(D_ * FF_) / (4 * 256), 256, 0, stream>>>(down_w, wsum, wq);

  dim3 g1(M_ / 128, FF_ / 64);
  k_gemm1_geglu<<<g1, 256, 0, stream>>>(hbf, upw_bf, hg);

  k_rowquant<<<M_, 256, 0, stream>>>(hg, hq, inv_a);

  dim3 g2(M_ / 128, D_ / 64);
  k_gemm2_bitlinear<<<g2, 256, 0, stream>>>(hq, wq, inv_a, wsum, x, out);
}
